// WordLSTM_31645319037043
// MI455X (gfx1250) — matrix operand order repaired and checked by reading, compiled, not run on hardware
//
#include <hip/hip_runtime.h>
#include <hip/hip_bf16.h>

typedef __attribute__((ext_vector_type(16))) __bf16 v16bf;
typedef __attribute__((ext_vector_type(8)))  float  v8f;

#define VOCAB  30000
#define EMBED  1024
#define HIDDEN 1024
#define BATCH  256
#define SEQ    128
#define TSTEPS (SEQ + 1)     // 129 (topic prepended)
#define G4     (4 * HIDDEN)  // 4096

static __device__ __forceinline__ unsigned short f2bf(float f) {
  unsigned int u = __float_as_uint(f);
  u += 0x7fffu + ((u >> 16) & 1u);   // round to nearest even
  return (unsigned short)(u >> 16);
}
static __device__ __forceinline__ float bf2f(unsigned short h) {
  return __uint_as_float(((unsigned int)h) << 16);
}

// ---------------- helper kernels ----------------

__global__ void f32_to_bf16_kernel(const float* __restrict__ src,
                                   unsigned short* __restrict__ dst, int n4) {
  int i = blockIdx.x * blockDim.x + threadIdx.x;
  if (i >= n4) return;
  float4 v = ((const float4*)src)[i];
  union { unsigned short u[4]; uint2 p; } o;
  o.u[0] = f2bf(v.x); o.u[1] = f2bf(v.y); o.u[2] = f2bf(v.z); o.u[3] = f2bf(v.w);
  ((uint2*)dst)[i] = o.p;
}

__global__ void bias_sum_kernel(const float* __restrict__ bi,
                                const float* __restrict__ bh,
                                float* __restrict__ out) {
  int i = blockIdx.x * blockDim.x + threadIdx.x;
  if (i < G4) out[i] = bi[i] + bh[i];
}

// Build x[t, b, e] (bf16): t==0 -> topic[b], else embed_w[captions[b][t-1]]
__global__ void gather_kernel(const float* __restrict__ topic,
                              const int* __restrict__ caps,
                              const float* __restrict__ emb,
                              unsigned short* __restrict__ x) {
  int idx = blockIdx.x * blockDim.x + threadIdx.x;   // over TSTEPS*BATCH*(EMBED/4)
  int e4 = idx & 255;
  int b  = (idx >> 8) & 255;
  int t  = idx >> 16;
  if (t >= TSTEPS) return;
  const float* row = (t == 0) ? (topic + (size_t)b * EMBED)
                              : (emb + (size_t)caps[b * SEQ + (t - 1)] * EMBED);
  float4 v = ((const float4*)row)[e4];
  union { unsigned short u[4]; uint2 p; } o;
  o.u[0] = f2bf(v.x); o.u[1] = f2bf(v.y); o.u[2] = f2bf(v.z); o.u[3] = f2bf(v.w);
  ((uint2*)x)[idx] = o.p;
}

__global__ void init_state_kernel(unsigned short* __restrict__ h,
                                  float* __restrict__ c) {
  int i = blockIdx.x * blockDim.x + threadIdx.x;
  if (i < BATCH * HIDDEN) { h[i] = 0; c[i] = 0.f; }
}

// LSTM cell elementwise: gates [B,4H] f32 (i,f,g,o), updates c (f32) and h (bf16)
__global__ void lstm_cell_kernel(const float* __restrict__ gates,
                                 float* __restrict__ c,
                                 unsigned short* __restrict__ h) {
  int i = blockIdx.x * blockDim.x + threadIdx.x;   // over B*H
  if (i >= BATCH * HIDDEN) return;
  int b  = i >> 10;
  int hh = i & 1023;
  const float* g = gates + (size_t)b * G4;
  float ig = g[hh];
  float fg = g[hh + HIDDEN];
  float gg = g[hh + 2 * HIDDEN];
  float og = g[hh + 3 * HIDDEN];
  ig = 1.f / (1.f + __expf(-ig));
  fg = 1.f / (1.f + __expf(-fg));
  og = 1.f / (1.f + __expf(-og));
  gg = tanhf(gg);
  float cn = fg * c[i] + ig * gg;
  c[i] = cn;
  h[i] = f2bf(og * tanhf(cn));
}

// ---------------- WMMA GEMM: Out[M,N] = A[M,K](bf16) @ B[N,K]^T(bf16) + Cinit ----
// CMODE:   0=zero init, 1=bias[n] (f32), 2=bf16 Cin[M,N] (pre-activations)
// OUTBF16: 1 -> bf16 output, 0 -> f32 output
// NCHECK:  true -> N not multiple of block tile (guard stores, clamp B row ptrs)
// KDIM:    compile-time K (== lda == ldb), multiple of 64
template<int CMODE, int OUTBF16, bool NCHECK, int KDIM>
__global__ __launch_bounds__(256) void gemm_bf16_wmma(
    const unsigned short* __restrict__ A,
    const unsigned short* __restrict__ Bw,
    const unsigned short* __restrict__ Cin,
    const float* __restrict__ bias,
    float* __restrict__ outF,
    unsigned short* __restrict__ outH,
    int M, int N)
{
  const int lane = threadIdx.x & 31;
  const int wave = threadIdx.x >> 5;   // 0..7, wave grid 2 (M) x 4 (N)
  const int wm   = wave >> 2;
  const int wn   = wave & 3;
  const int l15  = lane & 15;
  const int half = lane >> 4;          // wave32 half-lane select

  const int mW = blockIdx.y * 64  + wm * 32;   // wave tile: 32 x 64
  const int nW = blockIdx.x * 256 + wn * 64;

  // ---- accumulator init (C/D layout: VGPR r -> row rbase+r, col n0+l15) ----
  v8f acc[2][4];
  for (int mi = 0; mi < 2; ++mi)
    for (int ni = 0; ni < 4; ++ni) {
      const int col   = nW + ni * 16 + l15;
      const int rbase = mW + mi * 16 + half * 8;
      v8f a;
      for (int r = 0; r < 8; ++r) a[r] = 0.f;
      if (CMODE == 1) {
        float bv = (!NCHECK || col < N) ? bias[col] : 0.f;
        for (int r = 0; r < 8; ++r) a[r] = bv;
      } else if (CMODE == 2) {
        for (int r = 0; r < 8; ++r)
          a[r] = bf2f(Cin[(size_t)(rbase + r) * N + col]);
      }
      acc[mi][ni] = a;
    }

  // ---- pointer setup (clamp OOB columns to row 0: computed garbage never stored) ----
  const unsigned short* aBase = A + (size_t)(mW + l15) * KDIM + half * 8;
  const unsigned short* bBase[4];
  for (int ni = 0; ni < 4; ++ni) {
    int col = nW + ni * 16 + l15;
    if (NCHECK && col >= N) col = 0;
    bBase[ni] = Bw + (size_t)col * KDIM + half * 8;
  }

  // ---- software-pipelined K loop, fully unrolled (KDIM compile-time) ----
  union Frag { v16bf v; uint4 q[2]; };
  Frag af[2][2], bfr[2][4];

  auto loadK = [&](int buf, int k0) {
    for (int mi = 0; mi < 2; ++mi) {
      const unsigned short* p = aBase + (size_t)mi * 16 * KDIM + k0;
      af[buf][mi].q[0] = *(const uint4*)(p);        // K = k0+half*8 .. +7
      af[buf][mi].q[1] = *(const uint4*)(p + 16);   // K = +16 .. +23
    }
    for (int ni = 0; ni < 4; ++ni) {
      const unsigned short* p = bBase[ni] + k0;     // K = k0+half*8 .. +7, +16 .. +23
      bfr[buf][ni].q[0] = *(const uint4*)(p);
      bfr[buf][ni].q[1] = *(const uint4*)(p + 16);
    }
  };

  loadK(0, 0);
#pragma unroll
  for (int k0 = 0; k0 < KDIM; k0 += 32) {
    const int cur = (k0 >> 5) & 1;
    if (k0 + 32 < KDIM) loadK(cur ^ 1, k0 + 32);   // overlap next loads with WMMAs
    for (int mi = 0; mi < 2; ++mi)
      for (int ni = 0; ni < 4; ++ni)
        acc[mi][ni] = __builtin_amdgcn_wmma_f32_16x16x32_bf16(
            false, af[cur][mi].v, false, bfr[cur][ni].v,
            (short)0, acc[mi][ni], false, false);
  }

  // ---- epilogue ----
  for (int mi = 0; mi < 2; ++mi)
    for (int ni = 0; ni < 4; ++ni) {
      const int col = nW + ni * 16 + l15;
      if (NCHECK && col >= N) continue;
      const int rbase = mW + mi * 16 + half * 8;
      v8f a = acc[mi][ni];
      for (int r = 0; r < 8; ++r) {
        size_t o = (size_t)(rbase + r) * N + col;
        if (OUTBF16) outH[o] = f2bf(a[r]);
        else         outF[o] = a[r];
      }
    }
}

// ---------------- launch ----------------

extern "C" void kernel_launch(void* const* d_in, const int* in_sizes, int n_in,
                              void* d_out, int out_size, void* d_ws, size_t ws_size,
                              hipStream_t stream) {
  const float* topic = (const float*)d_in[0];
  const int*   caps  = (const int*)  d_in[1];
  const float* embw  = (const float*)d_in[2];
  const float* wih   = (const float*)d_in[3];
  const float* whh   = (const float*)d_in[4];
  const float* bih   = (const float*)d_in[5];
  const float* bhh   = (const float*)d_in[6];
  const float* linw  = (const float*)d_in[7];
  const float* linb  = (const float*)d_in[8];
  float* out = (float*)d_out;

  char* ws = (char*)d_ws;
  auto alloc = [&](size_t bytes) {
    char* p = ws;
    ws += (bytes + 255) & ~(size_t)255;
    return p;
  };
  unsigned short* x    = (unsigned short*)alloc((size_t)TSTEPS * BATCH * EMBED * 2);
  unsigned short* pre  = (unsigned short*)alloc((size_t)TSTEPS * BATCH * G4 * 2);
  unsigned short* wihB = (unsigned short*)alloc((size_t)G4 * EMBED * 2);
  unsigned short* whhB = (unsigned short*)alloc((size_t)G4 * HIDDEN * 2);
  unsigned short* linB = (unsigned short*)alloc((size_t)VOCAB * HIDDEN * 2);
  float*          bias4= (float*)         alloc((size_t)G4 * 4);
  float*          gates= (float*)         alloc((size_t)BATCH * G4 * 4);
  unsigned short* hbuf = (unsigned short*)alloc((size_t)BATCH * HIDDEN * 2);
  float*          cbuf = (float*)         alloc((size_t)BATCH * HIDDEN * 4);

  // weight conversions (f32 -> bf16), bias fuse, embedding gather, state init
  f32_to_bf16_kernel<<<(G4 * EMBED / 4 + 255) / 256, 256, 0, stream>>>(wih, wihB, G4 * EMBED / 4);
  f32_to_bf16_kernel<<<(G4 * HIDDEN / 4 + 255) / 256, 256, 0, stream>>>(whh, whhB, G4 * HIDDEN / 4);
  f32_to_bf16_kernel<<<(VOCAB * HIDDEN / 4 + 255) / 256, 256, 0, stream>>>(linw, linB, VOCAB * HIDDEN / 4);
  bias_sum_kernel<<<(G4 + 255) / 256, 256, 0, stream>>>(bih, bhh, bias4);
  gather_kernel<<<TSTEPS * BATCH * (EMBED / 4) / 256, 256, 0, stream>>>(topic, caps, embw, x);
  init_state_kernel<<<(BATCH * HIDDEN + 255) / 256, 256, 0, stream>>>(hbuf, cbuf);

  // Input GEMM over ALL timesteps at once: pre[T*B, 4H] = x @ w_ih^T + (b_ih+b_hh)
  {
    dim3 grid(G4 / 256, (TSTEPS * BATCH) / 64);
    gemm_bf16_wmma<1, 1, false, EMBED><<<grid, 256, 0, stream>>>(
        x, wihB, nullptr, bias4, nullptr, pre, TSTEPS * BATCH, G4);
  }

  // Sequential recurrence: gates = pre[t] + h @ w_hh^T ; then elementwise cell
  for (int t = 0; t < TSTEPS; ++t) {
    dim3 grid(G4 / 256, BATCH / 64);
    gemm_bf16_wmma<2, 0, false, HIDDEN><<<grid, 256, 0, stream>>>(
        hbuf, whhB, pre + (size_t)t * BATCH * G4, nullptr, gates, nullptr,
        BATCH, G4);
    lstm_cell_kernel<<<BATCH * HIDDEN / 256, 256, 0, stream>>>(gates, cbuf, hbuf);
  }

  // Decode: logits[B, V] = h_last @ lin_w^T + lin_b
  {
    dim3 grid((VOCAB + 255) / 256, BATCH / 64);
    gemm_bf16_wmma<1, 0, true, HIDDEN><<<grid, 256, 0, stream>>>(
        hbuf, linB, nullptr, linb, out, nullptr, BATCH, VOCAB);
  }
}
